// DriftAwareLightMemory_85744727097570
// MI455X (gfx1250) — matrix operand order repaired and checked by reading, not compiled, not run on hardware
//
#include <hip/hip_runtime.h>
#include <hip/hip_bf16.h>
#include <math.h>

// ---------- types ----------
typedef __attribute__((ext_vector_type(16))) __bf16 v16bf;
typedef __attribute__((ext_vector_type(8)))  __bf16 v8bf;
typedef __attribute__((ext_vector_type(8)))  float  v8f;
typedef __attribute__((ext_vector_type(4)))  float  f32x4;

// ---------- constants ----------
#define DD   512
#define BB   4
#define TT   16
#define LL   1024
#define ROWS 4096                 // B*L
#define WBLK 262144               // one 512x512 weight block (elements)

// ws byte offsets (all 256B aligned)
#define OFF_WT     0u             // 9 * 512*512 bf16      = 4,718,592 B
#define OFF_PRE    4718592u       // 4096*512 bf16         = 4,194,304 B
#define OFF_RAW    8912896u       // 4096*512 f32          = 8,388,608 B
#define OFF_ENH    17301504u      // 4096*512 f32          = 8,388,608 B
#define OFF_SM     25690112u      // 4*16*512 f32
#define OFF_PE     25821184u      // 16*512 f32
#define OFF_PEMB   25853952u      // 16*512 f32
#define OFF_DM     25886720u      // 4*512 f32
#define OFF_XRM    25894912u      // 4*512 f32
#define OFF_MG     25903104u      // 4*16*512 f32
#define OFF_MD     26034176u      // 4*16*512 f32
#define OFF_QG     26165248u      // 4*512 f32
#define OFF_CD     26173440u      // 4*512 f32
#define OFF_ATTN   26181632u      // 4*16 f32
#define OFF_PEMIX  26181888u      // 4*512 f32

// ---------- helpers ----------
// Native fp32 -> bf16 (backend emits v_cvt_pk_bf16_f32, RNE)
__device__ __forceinline__ __bf16 f2bf(float f) { return (__bf16)f; }

__device__ __forceinline__ v8f v8f_zero() {
  v8f z = {0.f,0.f,0.f,0.f,0.f,0.f,0.f,0.f}; return z;
}

__device__ __forceinline__ float sigmoidf_(float v) { return 1.f / (1.f + expf(-v)); }

__device__ __forceinline__ v8f wmma_bf16(v16bf a, v16bf b, v8f c) {
  return __builtin_amdgcn_wmma_f32_16x16x32_bf16(false, a, false, b, (short)0, c, false, false);
}

// A-fragment (16x32 bf16) from an fp32 row. Per ISA layout:
// lanes 0-15 hold K = [k0..k0+7] and [k0+16..k0+23]; lanes 16-31 the +8 chunks.
__device__ __forceinline__ v16bf load_a_f32(const float* rowp, int k0, int half) {
  const float* p0 = rowp + k0 + 8 * half;
  const float* p1 = p0 + 16;
  f32x4 a0 = *(const f32x4*)p0;      f32x4 a1 = *(const f32x4*)(p0 + 4);
  f32x4 a2 = *(const f32x4*)p1;      f32x4 a3 = *(const f32x4*)(p1 + 4);
  v16bf a;
#pragma unroll
  for (int j = 0; j < 4; ++j) {
    a[j]      = f2bf(a0[j]);  a[4 + j]  = f2bf(a1[j]);
    a[8 + j]  = f2bf(a2[j]);  a[12 + j] = f2bf(a3[j]);
  }
  return a;
}

// Three A-fragments at once: x, x_phys, and delta = x - x_phys (fp32 subtract, then cvt)
__device__ __forceinline__ void load_a3(const float* xr, const float* pr, int k0, int half,
                                        v16bf& ax, v16bf& ap, v16bf& ad) {
  const int c0 = k0 + 8 * half, c1 = c0 + 16;
  f32x4 x0 = *(const f32x4*)(xr + c0); f32x4 x1 = *(const f32x4*)(xr + c0 + 4);
  f32x4 x2 = *(const f32x4*)(xr + c1); f32x4 x3 = *(const f32x4*)(xr + c1 + 4);
  f32x4 p0 = *(const f32x4*)(pr + c0); f32x4 p1 = *(const f32x4*)(pr + c0 + 4);
  f32x4 p2 = *(const f32x4*)(pr + c1); f32x4 p3 = *(const f32x4*)(pr + c1 + 4);
#pragma unroll
  for (int j = 0; j < 4; ++j) {
    ax[j]      = f2bf(x0[j]); ap[j]      = f2bf(p0[j]); ad[j]      = f2bf(x0[j] - p0[j]);
    ax[4 + j]  = f2bf(x1[j]); ap[4 + j]  = f2bf(p1[j]); ad[4 + j]  = f2bf(x1[j] - p1[j]);
    ax[8 + j]  = f2bf(x2[j]); ap[8 + j]  = f2bf(p2[j]); ad[8 + j]  = f2bf(x2[j] - p2[j]);
    ax[12 + j] = f2bf(x3[j]); ap[12 + j] = f2bf(p3[j]); ad[12 + j] = f2bf(x3[j] - p3[j]);
  }
}

// A-fragment from a bf16 row
__device__ __forceinline__ v16bf load_a_bf(const __bf16* rowp, int k0, int half) {
  v8bf lo = *(const v8bf*)(rowp + k0 + 8 * half);
  v8bf hi = *(const v8bf*)(rowp + k0 + 16 + 8 * half);
  v16bf a;
#pragma unroll
  for (int j = 0; j < 8; ++j) { a[j] = lo[j]; a[8 + j] = hi[j]; }
  return a;
}

// B-fragment (32x16 bf16) from pre-transposed weight Wt[n][k]:
// lane m<16 -> column n=nbase+m, K=[k0..k0+7],[k0+16..k0+23]; lanes 16-31 the +8 chunks
__device__ __forceinline__ v16bf load_b(const __bf16* wt, int nbase, int k0, int lane) {
  int mm = lane & 15, hh = lane >> 4;
  const __bf16* p = wt + (size_t)(nbase + mm) * DD + k0 + 8 * hh;
  v8bf lo = *(const v8bf*)p;
  v8bf hi = *(const v8bf*)(p + 16);
  v16bf b;
#pragma unroll
  for (int j = 0; j < 8; ++j) { b[j] = lo[j]; b[8 + j] = hi[j]; }
  return b;
}

// ---------- kernel 1: weight prep (transpose + bf16; phys pre-negated) ----------
// wt blocks: 0=delta 1=xproj 2=-phys 3=gate[0:D] 4=gate[D:2D] 5=gate[2D:3D] 6=outp 7=fuse[0:D] 8=fuse[D:2D]
__global__ __launch_bounds__(256) void k_prep(const float* dW, const float* xW, const float* pW,
                                              const float* gW, const float* oW, const float* fW,
                                              __bf16* wt) {
  int idx = blockIdx.x * 256 + threadIdx.x;
  if (idx >= 9 * WBLK) return;
  int w = idx >> 18;
  int rem = idx & (WBLK - 1);
  int k = rem >> 9;
  int n = rem & 511;
  float v;
  switch (w) {
    case 0:  v =  dW[(size_t)k * DD + n]; break;
    case 1:  v =  xW[(size_t)k * DD + n]; break;
    case 2:  v = -pW[(size_t)k * DD + n]; break;
    case 3:  v =  gW[(size_t)k * DD + n]; break;
    case 4:  v =  gW[(size_t)(k + 512) * DD + n]; break;
    case 5:  v =  gW[(size_t)(k + 1024) * DD + n]; break;
    case 6:  v =  oW[(size_t)k * DD + n]; break;
    case 7:  v =  fW[(size_t)k * DD + n]; break;
    default: v =  fW[(size_t)(k + 512) * DD + n]; break;
  }
  wt[(size_t)w * WBLK + (size_t)n * DD + k] = f2bf(v);
}

// ---------- kernel 2: fused stage-1 WMMA GEMMs -> pre (bf16) ----------
// pre = g*delta_feat + (1-g)*(x_feat - phys_feat), with g = sigmoid(gate)
__global__ __launch_bounds__(256) void k_stage1(const float* __restrict__ x,
                                                const float* __restrict__ snap,
                                                const __bf16* __restrict__ wt,
                                                const float* delta_b, const float* xproj_b,
                                                const float* phys_b, const float* gate_b,
                                                __bf16* __restrict__ pre) {
  const int lane = threadIdx.x & 31;
  const int wave = threadIdx.x >> 5;
  const int rowBase = (blockIdx.x * 8 + wave) * 16;   // grid.x = 32
  const int colBase = blockIdx.y * 64;                // grid.y = 8
  const int half = lane >> 4;
  const int m = lane & 15;
  const int row = rowBase + m;
  const float* xr = x + (size_t)row * DD;
  const float* pr = snap + (((size_t)(row >> 10) * TT + (TT - 1)) * LL + (row & (LL - 1))) * DD;

  const __bf16* Wd  = wt + 0 * (size_t)WBLK;
  const __bf16* Wx  = wt + 1 * (size_t)WBLK;
  const __bf16* Wpn = wt + 2 * (size_t)WBLK;   // -phys_W
  const __bf16* G0  = wt + 3 * (size_t)WBLK;
  const __bf16* G1  = wt + 4 * (size_t)WBLK;
  const __bf16* G2  = wt + 5 * (size_t)WBLK;

  v8f df[4], uu[4], gs[4];
#pragma unroll
  for (int t = 0; t < 4; ++t) { df[t] = v8f_zero(); uu[t] = v8f_zero(); gs[t] = v8f_zero(); }

  for (int k0 = 0; k0 < DD; k0 += 32) {
    v16bf ax, ap, ad;
    load_a3(xr, pr, k0, half, ax, ap, ad);
#pragma unroll
    for (int t = 0; t < 4; ++t) {
      const int nt = colBase + t * 16;
      v16bf b;
      b = load_b(Wd,  nt, k0, lane);  df[t] = wmma_bf16(ad, b, df[t]);  // delta @ delta_W
      b = load_b(Wx,  nt, k0, lane);  uu[t] = wmma_bf16(ax, b, uu[t]);  // x @ xproj_W
      b = load_b(Wpn, nt, k0, lane);  uu[t] = wmma_bf16(ap, b, uu[t]);  // - x_phys @ phys_W
      b = load_b(G0,  nt, k0, lane);  gs[t] = wmma_bf16(ad, b, gs[t]);  // delta @ gW0
      b = load_b(G1,  nt, k0, lane);  gs[t] = wmma_bf16(ax, b, gs[t]);  // x @ gW1
      b = load_b(G2,  nt, k0, lane);  gs[t] = wmma_bf16(ap, b, gs[t]);  // x_phys @ gW2
    }
  }

#pragma unroll
  for (int t = 0; t < 4; ++t) {
    const int n = colBase + t * 16 + m;
    const float db = delta_b[n];
    const float ub = xproj_b[n] - phys_b[n];
    const float gb = gate_b[n];
#pragma unroll
    for (int v = 0; v < 8; ++v) {
      const int rr = rowBase + v + 8 * half;
      const float dfe = df[t][v] + db;
      const float ue  = uu[t][v] + ub;
      const float g   = sigmoidf_(gs[t][v] + gb);
      pre[(size_t)rr * DD + n] = f2bf(g * dfe + (1.f - g) * ue);
    }
  }
}

// ---------- kernel 3: raw = pre @ outp_W + outp_b (WMMA) ----------
__global__ __launch_bounds__(256) void k_gemm_outp(const __bf16* __restrict__ pre,
                                                   const __bf16* __restrict__ wt6,
                                                   const float* outp_b,
                                                   float* __restrict__ raw) {
  const int lane = threadIdx.x & 31;
  const int wave = threadIdx.x >> 5;
  const int rowBase = (blockIdx.x * 8 + wave) * 16;   // grid.x = 32
  const int colBase = blockIdx.y * 128;               // grid.y = 4
  const int half = lane >> 4;
  const int m = lane & 15;
  const __bf16* ar = pre + (size_t)(rowBase + m) * DD;

  v8f acc[8];
#pragma unroll
  for (int t = 0; t < 8; ++t) acc[t] = v8f_zero();

#pragma unroll 2
  for (int k0 = 0; k0 < DD; k0 += 32) {
    v16bf a = load_a_bf(ar, k0, half);
#pragma unroll
    for (int t = 0; t < 8; ++t) {
      v16bf b = load_b(wt6, colBase + t * 16, k0, lane);
      acc[t] = wmma_bf16(a, b, acc[t]);
    }
  }
#pragma unroll
  for (int t = 0; t < 8; ++t) {
    const int n = colBase + t * 16 + m;
    const float ob = outp_b[n];
#pragma unroll
    for (int v = 0; v < 8; ++v) {
      const int rr = rowBase + v + 8 * half;
      raw[(size_t)rr * DD + n] = acc[t][v] + ob;
    }
  }
}

// ---------- kernel 4: snap_mean[b,t,d] = mean_l snapshot ----------
__global__ __launch_bounds__(256) void k_snapmean(const float* __restrict__ snap, float* sm) {
  int idx = blockIdx.x * 256 + threadIdx.x;          // 32768 = B*T*D
  int d = idx & 511;
  int bt = idx >> 9;
  const float* p = snap + (size_t)bt * LL * DD + d;
  float s = 0.f;
  for (int l = 0; l < LL; ++l) {
    if (l + 8 < LL) __builtin_prefetch(p + (size_t)(l + 8) * DD, 0, 1);
    s += p[(size_t)l * DD];
  }
  sm[idx] = s * (1.0f / (float)LL);
}

// ---------- kernel 5: delta_mean + xr_mean over L ----------
__global__ __launch_bounds__(256) void k_rowmeans(const float* __restrict__ x,
                                                  const float* __restrict__ snap,
                                                  const float* __restrict__ raw,
                                                  float* dmean, float* xrmean) {
  int idx = blockIdx.x * 256 + threadIdx.x;          // 2048 = B*D
  int b = idx >> 9;
  int d = idx & 511;
  const float* xp = x    + (size_t)b * LL * DD + d;
  const float* pp = snap + ((size_t)b * TT + (TT - 1)) * LL * DD + d;
  const float* rp = raw  + (size_t)b * LL * DD + d;
  float sd = 0.f, sx = 0.f;
  for (int l = 0; l < LL; ++l) {
    float xv = xp[(size_t)l * DD];
    sd += xv - pp[(size_t)l * DD];
    sx += xv + rp[(size_t)l * DD];
  }
  dmean[idx]  = sd * (1.0f / (float)LL);
  xrmean[idx] = sx * (1.0f / (float)LL);
}

// ---------- kernel 6: sinusoid table pe[t][d] ----------
__global__ __launch_bounds__(256) void k_pe(float* pe) {
  for (int w = threadIdx.x; w < TT * 256; w += 256) {
    int t = w >> 8;
    int i = w & 255;
    float pos = (float)(t + 1);
    float div = expf(-9.210340371976184f * (2.0f * (float)i) / (float)DD);  // ln(1e4)
    float ang = pos * div;
    pe[t * DD + 2 * i]     = sinf(ang);
    pe[t * DD + 2 * i + 1] = cosf(ang);
  }
}

// ---------- kernel 7: pos_emb = pe @ seq_W + seq_b ----------
__global__ __launch_bounds__(256) void k_posemb(const float* __restrict__ pe,
                                                const float* __restrict__ seq_W,
                                                const float* seq_b, float* pemb) {
  int t = blockIdx.x;
  __shared__ float src[DD];
  for (int d = threadIdx.x; d < DD; d += 256) src[d] = pe[t * DD + d];
  __syncthreads();
  for (int n = threadIdx.x; n < DD; n += 256) {
    float s = seq_b[n];
    for (int k = 0; k < DD; ++k) s += src[k] * seq_W[(size_t)k * DD + n];
    pemb[t * DD + n] = s;
  }
}

// ---------- kernel 8: batched small matvecs ----------
// blocks 0..63: memory_global; 64..127: mem_drift; 128..131: q_global; 132..135: cur_drift
__global__ __launch_bounds__(256) void k_matvecs(const float* __restrict__ sm,
                                                 const float* __restrict__ pemb,
                                                 const float* __restrict__ xrm,
                                                 const float* __restrict__ dmean,
                                                 const float* mem_W, const float* mem_b,
                                                 const float* memd_W, const float* memd_b,
                                                 const float* q_W, const float* q_b,
                                                 const float* curd_W, const float* curd_b,
                                                 float* mg, float* md, float* qg, float* cd) {
  __shared__ float src[DD];
  int blk = blockIdx.x;
  const float* W; const float* bias; float* out;
  if (blk < 64) {
    int bt = blk, t = bt & 15;
    for (int d = threadIdx.x; d < DD; d += 256)
      src[d] = sm[bt * DD + d] + pemb[t * DD + d];
    W = mem_W; bias = mem_b; out = mg + bt * DD;
  } else if (blk < 128) {
    int bt = blk - 64, t = bt & 15;
    for (int d = threadIdx.x; d < DD; d += 256) {
      float v;
      if (t == 0) v = sm[bt * DD + d] + pemb[d];
      else v = sm[bt * DD + d] - sm[(bt - 1) * DD + d]
             + pemb[t * DD + d] - pemb[(t - 1) * DD + d];
      src[d] = v;
    }
    W = memd_W; bias = memd_b; out = md + bt * DD;
  } else if (blk < 132) {
    int b = blk - 128;
    for (int d = threadIdx.x; d < DD; d += 256) src[d] = xrm[b * DD + d];
    W = q_W; bias = q_b; out = qg + b * DD;
  } else {
    int b = blk - 132;
    for (int d = threadIdx.x; d < DD; d += 256) src[d] = dmean[b * DD + d];
    W = curd_W; bias = curd_b; out = cd + b * DD;
  }
  __syncthreads();
  for (int n = threadIdx.x; n < DD; n += 256) {
    float s = bias[n];
    for (int k = 0; k < DD; ++k) s += src[k] * W[(size_t)k * DD + n];
    out[n] = s;
  }
}

// ---------- kernel 9: scores, softmax, attn, pe_mix ----------
__global__ __launch_bounds__(256) void k_attn(const float* __restrict__ qg,
                                              const float* __restrict__ mg,
                                              const float* __restrict__ cd,
                                              const float* __restrict__ md,
                                              const float* __restrict__ pemb,
                                              float* attn, float* pemix) {
  int b = blockIdx.x;
  __shared__ float sc[TT], at[TT];
  int tid = threadIdx.x;
  if (tid < TT) {
    const float* q  = qg + b * DD;
    const float* mm = mg + (b * TT + tid) * DD;
    const float* c  = cd + b * DD;
    const float* dp = md + (b * TT + tid) * DD;
    float dc = 0.f, ds = 0.f;
    for (int k = 0; k < DD; ++k) {
      dc += q[k] * mm[k];
      float dd = c[k] - dp[k];
      ds += dd * dd;
    }
    sc[tid] = dc * 0.04419417382415922f - 0.3f * (ds * (1.f / (float)DD));
  }
  __syncthreads();
  if (tid == 0) {
    float mx = sc[0];
    for (int t = 1; t < TT; ++t) mx = fmaxf(mx, sc[t]);
    float s = 0.f;
    for (int t = 0; t < TT; ++t) { at[t] = expf(sc[t] - mx); s += at[t]; }
    float inv = 1.f / s;
    for (int t = 0; t < TT; ++t) { at[t] *= inv; attn[b * TT + t] = at[t]; }
  }
  __syncthreads();
  for (int d = tid; d < DD; d += 256) {
    float s = 0.f;
    for (int t = 0; t < TT; ++t) s += at[t] * pemb[t * DD + d];
    pemix[b * DD + d] = s;
  }
}

// ---------- kernel 10: enhanced = pe_mix + sum_t attn * snapshot ----------
__global__ __launch_bounds__(256) void k_enh(const float* __restrict__ snap,
                                             const float* __restrict__ attn,
                                             const float* __restrict__ pemix,
                                             float* __restrict__ enh) {
  size_t idx = (size_t)blockIdx.x * 256 + threadIdx.x;  // B*L*D
  int d = (int)(idx & 511);
  size_t bl = idx >> 9;
  int b = (int)(bl >> 10);
  const float* p = snap + (size_t)b * TT * LL * DD + (bl & (LL - 1)) * DD + d;
  float s = pemix[b * DD + d];
#pragma unroll
  for (int t = 0; t < TT; ++t)
    s += attn[b * TT + t] * p[(size_t)t * LL * DD];
  enh[idx] = s;
}

// ---------- kernel 11: fuse GEMM + final residual output (WMMA) ----------
__global__ __launch_bounds__(256) void k_fuse(const float* __restrict__ x,
                                              const float* __restrict__ enh,
                                              const float* __restrict__ raw,
                                              const __bf16* __restrict__ wtF0,
                                              const __bf16* __restrict__ wtF1,
                                              const float* fuse_b,
                                              float* __restrict__ out) {
  const int lane = threadIdx.x & 31;
  const int wave = threadIdx.x >> 5;
  const int rowBase = (blockIdx.x * 8 + wave) * 16;   // grid.x = 32
  const int colBase = blockIdx.y * 64;                // grid.y = 8
  const int half = lane >> 4;
  const int m = lane & 15;
  const float* xr = x   + (size_t)(rowBase + m) * DD;
  const float* er = enh + (size_t)(rowBase + m) * DD;

  v8f acc[4];
#pragma unroll
  for (int t = 0; t < 4; ++t) acc[t] = v8f_zero();

#pragma unroll 2
  for (int k0 = 0; k0 < DD; k0 += 32) {
    v16bf axf = load_a_f32(xr, k0, half);
    v16bf aef = load_a_f32(er, k0, half);
#pragma unroll
    for (int t = 0; t < 4; ++t) {
      const int nt = colBase + t * 16;
      v16bf b0 = load_b(wtF0, nt, k0, lane);
      acc[t] = wmma_bf16(axf, b0, acc[t]);
      v16bf b1 = load_b(wtF1, nt, k0, lane);
      acc[t] = wmma_bf16(aef, b1, acc[t]);
    }
  }
#pragma unroll
  for (int t = 0; t < 4; ++t) {
    const int n = colBase + t * 16 + m;
    const float fb = fuse_b[n];
#pragma unroll
    for (int v = 0; v < 8; ++v) {
      const int rr = rowBase + v + 8 * half;
      const size_t ix = (size_t)rr * DD + n;
      const float g = sigmoidf_(acc[t][v] + fb);
      out[ix] = x[ix] + raw[ix] + g * enh[ix];
    }
  }
}

// ---------- host ----------
extern "C" void kernel_launch(void* const* d_in, const int* in_sizes, int n_in,
                              void* d_out, int out_size, void* d_ws, size_t ws_size,
                              hipStream_t stream) {
  const float* x       = (const float*)d_in[0];
  const float* snap    = (const float*)d_in[1];
  const float* delta_W = (const float*)d_in[2];  const float* delta_b = (const float*)d_in[3];
  const float* xproj_W = (const float*)d_in[4];  const float* xproj_b = (const float*)d_in[5];
  const float* phys_W  = (const float*)d_in[6];  const float* phys_b  = (const float*)d_in[7];
  const float* gate_W  = (const float*)d_in[8];  const float* gate_b  = (const float*)d_in[9];
  const float* outp_W  = (const float*)d_in[10]; const float* outp_b  = (const float*)d_in[11];
  const float* seq_W   = (const float*)d_in[12]; const float* seq_b   = (const float*)d_in[13];
  const float* q_W     = (const float*)d_in[14]; const float* q_b     = (const float*)d_in[15];
  const float* mem_W   = (const float*)d_in[16]; const float* mem_b   = (const float*)d_in[17];
  const float* curd_W  = (const float*)d_in[18]; const float* curd_b  = (const float*)d_in[19];
  const float* memd_W  = (const float*)d_in[20]; const float* memd_b  = (const float*)d_in[21];
  const float* fuse_W  = (const float*)d_in[22]; const float* fuse_b  = (const float*)d_in[23];

  char* ws = (char*)d_ws;
  __bf16* wt   = (__bf16*)(ws + OFF_WT);
  __bf16* pre  = (__bf16*)(ws + OFF_PRE);
  float* raw   = (float*)(ws + OFF_RAW);
  float* enh   = (float*)(ws + OFF_ENH);
  float* sm    = (float*)(ws + OFF_SM);
  float* pe    = (float*)(ws + OFF_PE);
  float* pemb  = (float*)(ws + OFF_PEMB);
  float* dmean = (float*)(ws + OFF_DM);
  float* xrm   = (float*)(ws + OFF_XRM);
  float* mg    = (float*)(ws + OFF_MG);
  float* md    = (float*)(ws + OFF_MD);
  float* qg    = (float*)(ws + OFF_QG);
  float* cd    = (float*)(ws + OFF_CD);
  float* attn  = (float*)(ws + OFF_ATTN);
  float* pemix = (float*)(ws + OFF_PEMIX);
  float* out   = (float*)d_out;

  // 1. weight transpose/convert (phys pre-negated)
  k_prep<<<9 * WBLK / 256, 256, 0, stream>>>(delta_W, xproj_W, phys_W, gate_W, outp_W, fuse_W, wt);
  // 2. fused stage-1 GEMMs -> pre
  k_stage1<<<dim3(32, 8), 256, 0, stream>>>(x, snap, wt, delta_b, xproj_b, phys_b, gate_b, pre);
  // 3. raw_memory = pre @ outp_W + outp_b
  k_gemm_outp<<<dim3(32, 4), 256, 0, stream>>>(pre, wt + 6 * (size_t)WBLK, outp_b, raw);
  // 4. snapshot mean over L
  k_snapmean<<<BB * TT * DD / 256, 256, 0, stream>>>(snap, sm);
  // 5. delta_mean + (x+raw) mean over L
  k_rowmeans<<<BB * DD / 256, 256, 0, stream>>>(x, snap, raw, dmean, xrm);
  // 6. sinusoid table
  k_pe<<<1, 256, 0, stream>>>(pe);
  // 7. pos_emb
  k_posemb<<<TT, 256, 0, stream>>>(pe, seq_W, seq_b, pemb);
  // 8. memory_global / mem_drift / q_global / cur_drift
  k_matvecs<<<136, 256, 0, stream>>>(sm, pemb, xrm, dmean,
                                     mem_W, mem_b, memd_W, memd_b,
                                     q_W, q_b, curd_W, curd_b,
                                     mg, md, qg, cd);
  // 9. scores + softmax + pe_mix
  k_attn<<<BB, 256, 0, stream>>>(qg, mg, cd, md, pemb, attn, pemix);
  // 10. enhanced (second full snapshot pass)
  k_enh<<<ROWS * DD / 256, 256, 0, stream>>>(snap, attn, pemix, enh);
  // 11. fuse-gate GEMM + final output
  k_fuse<<<dim3(32, 8), 256, 0, stream>>>(x, enh, raw, wt + 7 * (size_t)WBLK,
                                          wt + 8 * (size_t)WBLK, fuse_b, out);
}
